// StudentQAT_67345087201812
// MI455X (gfx1250) — 8-bit matrix operand order repaired and checked by reading, compiled, not run on hardware
//
#include <hip/hip_runtime.h>
#include <hip/hip_bf16.h>
#include <stdint.h>

typedef int v8i __attribute__((ext_vector_type(8)));

#define NIMG 8192
#define NWAVES 8

#if __has_builtin(__builtin_amdgcn_global_load_async_to_lds_b128) && \
    __has_builtin(__builtin_amdgcn_s_wait_asynccnt)
#define HAS_ASYNC_LDS 1
typedef int v4i_gcc __attribute__((vector_size(16)));
typedef __attribute__((address_space(1))) v4i_gcc* gv4p;
typedef __attribute__((address_space(3))) v4i_gcc* lv4p;
#else
#define HAS_ASYNC_LDS 0
#endif

static constexpr size_t WS_ABSMAX = 0;
static constexpr size_t WS_A1   = 256;
static constexpr size_t WS_A2   = WS_A1  + (size_t)NIMG * 16 * 676;
static constexpr size_t WS_A3   = WS_A2  + (size_t)NIMG * 16 * 576;
static constexpr size_t WS_POOL = WS_A3  + (size_t)NIMG * 32 * 484;
static constexpr size_t WS_BF2  = WS_POOL + (size_t)NIMG * 32 * 121;
static constexpr size_t WS_BF3  = WS_BF2 + 4096;
static constexpr size_t WS_WFC  = WS_BF3 + 8192;
static constexpr size_t WS_END  = WS_WFC + 39168;

__global__ void k_init(unsigned* absmax) { *absmax = 0u; }

__global__ void k_conv1_absmax(const float* __restrict__ x,
                               const float* __restrict__ w1,
                               unsigned* __restrict__ absmax) {
    __shared__ float smax[256];
    const size_t idx = (size_t)blockIdx.x * blockDim.x + threadIdx.x;
    float lm = 0.f;
    if (idx < (size_t)NIMG * 676) {
        const int i  = (int)idx;
        const int b  = i / 676;
        const int p  = i - b * 676;
        const int oy = p / 26, ox = p - (p / 26) * 26;
        const float* xi = x + (size_t)b * 784 + oy * 28 + ox;
        float win[9];
#pragma unroll
        for (int ky = 0; ky < 3; ++ky)
#pragma unroll
            for (int kx = 0; kx < 3; ++kx) win[ky * 3 + kx] = xi[ky * 28 + kx];
#pragma unroll
        for (int co = 0; co < 16; ++co) {
            float acc = 0.f;
#pragma unroll
            for (int j = 0; j < 9; ++j) acc = fmaf(win[j], w1[co * 9 + j], acc);
            lm = fmaxf(lm, fabsf(acc));
        }
    }
    smax[threadIdx.x] = lm;
    __syncthreads();
    for (int s = 128; s > 0; s >>= 1) {
        if ((int)threadIdx.x < s) smax[threadIdx.x] = fmaxf(smax[threadIdx.x], smax[threadIdx.x + s]);
        __syncthreads();
    }
    if (threadIdx.x == 0) atomicMax(absmax, __float_as_uint(smax[0]));
}

__global__ void k_conv1_sign(const float* __restrict__ x,
                             const float* __restrict__ w1,
                             const unsigned* __restrict__ absmax,
                             int8_t* __restrict__ a1) {
    const size_t idx = (size_t)blockIdx.x * blockDim.x + threadIdx.x;
    if (idx >= (size_t)NIMG * 676) return;
    const float scale = __uint_as_float(*absmax) / 127.f + 1e-8f;
    const float inv   = 1.f / scale;
    const int i  = (int)idx;
    const int b  = i / 676;
    const int p  = i - b * 676;
    const int oy = p / 26, ox = p - (p / 26) * 26;
    const float* xi = x + (size_t)b * 784 + oy * 28 + ox;
    float win[9];
#pragma unroll
    for (int ky = 0; ky < 3; ++ky)
#pragma unroll
        for (int kx = 0; kx < 3; ++kx) win[ky * 3 + kx] = xi[ky * 28 + kx];
    int8_t* o = a1 + (size_t)b * (16 * 676) + p;
#pragma unroll
    for (int co = 0; co < 16; ++co) {
        float acc = 0.f;
#pragma unroll
        for (int j = 0; j < 9; ++j) acc = fmaf(win[j], w1[co * 9 + j], acc);
        const float q = rintf(acc * inv);
        o[co * 676] = (int8_t)((q > 0.f) - (q < 0.f));
    }
}

template <int CI, int CO>
__global__ void k_pack_bfrag(const float* __restrict__ w, int* __restrict__ bfrag) {
    constexpr int KTOT = CI * 9;
    constexpr int NT   = CO / 16;
    constexpr int TOT  = NT * 3 * 8 * 32;
    const int id = blockIdx.x * blockDim.x + threadIdx.x;
    if (id >= TOT) return;
    const int lane  = id & 31;
    const int v     = (id >> 5) & 7;
    const int rest  = id >> 8;
    const int chunk = rest % 3;
    const int t     = rest / 3;
    const int half  = lane >> 4;
    const int n     = (lane & 15) + t * 16;
    unsigned word = 0;
#pragma unroll
    for (int bb = 0; bb < 4; ++bb) {
        const int k = chunk * 64 + (v >> 2) * 32 + half * 16 + (v & 3) * 4 + bb;
        int s = 0;
        if (k < KTOT) {
            const float wv = w[(size_t)n * KTOT + k];
            s = (wv > 0.f) - (wv < 0.f);
        }
        word |= (unsigned)(s & 0xFF) << (8 * bb);
    }
    bfrag[id] = (int)word;
}

__global__ void k_pack_wfc(const float* __restrict__ wfc, int8_t* __restrict__ out) {
    const int i = blockIdx.x * blockDim.x + threadIdx.x;
    if (i >= 10 * 3872) return;
    const float v = wfc[i];
    out[i] = (int8_t)((v > 0.f) - (v < 0.f));
}

template <int CI, int CO, int HIN, int HOUT>
__global__ void __launch_bounds__(256) k_bin_conv_wmma(const int8_t* __restrict__ ain,
                                                       const int* __restrict__ bfrag,
                                                       int8_t* __restrict__ aout) {
    constexpr int PIX      = HOUT * HOUT;
    constexpr int TILES    = (PIX + 15) / 16;
    constexpr int KTOT     = CI * 9;
    constexpr int CH       = HIN * HIN;
    constexpr int INSTRIDE = CI * CH;
    constexpr int NT       = CO / 16;

    __shared__ alignas(16) int8_t limg[INSTRIDE];

    const int img  = blockIdx.x;
    const int tid  = threadIdx.x;
    const int lane = tid & 31;
    const int wid  = tid >> 5;
    const int half = lane >> 4;
    const int m    = lane & 15;

    {
        const int8_t* gsrc = ain + (size_t)img * INSTRIDE;
#if HAS_ASYNC_LDS
#pragma unroll 1
        for (int i = tid; i < INSTRIDE / 16; i += 256) {
            __builtin_amdgcn_global_load_async_to_lds_b128(
                (gv4p)(gsrc + i * 16), (lv4p)(limg + i * 16), 0, 0);
        }
        __builtin_amdgcn_s_wait_asynccnt(0);
#else
        const int4* gs = (const int4*)gsrc;
        int4* ld = (int4*)limg;
#pragma unroll 1
        for (int i = tid; i < INSTRIDE / 16; i += 256) ld[i] = gs[i];
#endif
    }
    __syncthreads();

    v8i bmr[NT * 3];
#pragma unroll
    for (int f = 0; f < NT * 3; ++f)
#pragma unroll
        for (int v = 0; v < 8; ++v)
            bmr[f][v] = bfrag[(f * 8 + v) * 32 + lane];

    auto AOFF = [](int k) {
        const int ci = k / 9;
        const int r  = k - ci * 9;
        const int ky = r / 3;
        return ci * CH + ky * HIN + (r - ky * 3);
    };

#pragma unroll 1
    for (int tile = wid; tile < TILES; tile += NWAVES) {
        const int p0 = tile * 16;
        int p = p0 + m;
        if (p >= PIX) p = PIX - 1;
        const int y     = p / HOUT;
        const int lbase = y * HIN + (p - y * HOUT);

        v8i acc[NT];
#pragma unroll
        for (int t = 0; t < NT; ++t)
#pragma unroll
            for (int r = 0; r < 8; ++r) acc[t][r] = 0;

#pragma unroll
        for (int chunk = 0; chunk < 3; ++chunk) {
            v8i a;
#pragma unroll
            for (int v = 0; v < 8; ++v) {
                unsigned w = 0;
#pragma unroll
                for (int bb = 0; bb < 4; ++bb) {
                    const int k0 = chunk * 64 + (v >> 2) * 32 + (v & 3) * 4 + bb;
                    const int k1 = k0 + 16;
                    unsigned byte = 0;
                    if (k1 < KTOT) {
                        const int off = half ? AOFF(k1) : AOFF(k0);
                        byte = (unsigned)(unsigned char)limg[lbase + off];
                    } else if (k0 < KTOT) {
                        byte = (unsigned)(unsigned char)limg[lbase + AOFF(k0)] &
                               (half ? 0u : 0xFFu);
                    }
                    w |= byte << (8 * bb);
                }
                a[v] = (int)w;
            }
#pragma unroll
            for (int t = 0; t < NT; ++t) {
                acc[t] = __builtin_amdgcn_wmma_i32_16x16x64_iu8(
                     true, a,  true, bmr[t * 3 + chunk], acc[t],
                     false,  false);
            }
        }

        const int ncol = lane & 15;
#pragma unroll
        for (int t = 0; t < NT; ++t) {
            const int n = ncol + t * 16;
            int8_t* outp = aout + (size_t)img * (CO * PIX) + (size_t)n * PIX;
#pragma unroll
            for (int r = 0; r < 8; ++r) {
                const int pp = p0 + r + half * 8;
                if (pp < PIX) {
                    const int s = acc[t][r];
                    outp[pp] = (int8_t)((s > 0) - (s < 0));
                }
            }
        }
    }
}

__global__ void k_pool(const int8_t* __restrict__ a3, int8_t* __restrict__ pool) {
    const size_t idx = (size_t)blockIdx.x * blockDim.x + threadIdx.x;
    if (idx >= (size_t)NIMG * 32 * 121) return;
    const int i  = (int)(idx % ((size_t)32 * 121));
    const int b  = (int)(idx / ((size_t)32 * 121));
    const int c  = i / 121;
    const int q  = i - c * 121;
    const int py = q / 11, px = q - (q / 11) * 11;
    const int8_t* s = a3 + ((size_t)b * 32 + c) * 484 + (2 * py) * 22 + 2 * px;
    const int sum = (int)s[0] + (int)s[1] + (int)s[22] + (int)s[23];
    pool[idx] = (int8_t)sum;
}

__global__ void k_fc(const int8_t* __restrict__ pool, const int8_t* __restrict__ wfc,
                     const float* __restrict__ bias, float* __restrict__ out) {
    const int idx = blockIdx.x * blockDim.x + threadIdx.x;
    if (idx >= NIMG * 10) return;
    const int n = idx % 10;
    const int b = idx / 10;
    const int8_t* s = pool + (size_t)b * 3872;
    const int8_t* w = wfc + (size_t)n * 3872;
    int acc = 0;
    for (int k = 0; k < 3872; k += 8) {
#pragma unroll
        for (int j = 0; j < 8; ++j) acc += (int)s[k + j] * (int)w[k + j];
    }
    out[idx] = 0.25f * (float)acc + bias[n];
}

extern "C" void kernel_launch(void* const* d_in, const int* in_sizes, int n_in,
                              void* d_out, int out_size, void* d_ws, size_t ws_size,
                              hipStream_t stream) {
    (void)in_sizes; (void)n_in; (void)out_size;
    if (ws_size < WS_END) return;

    const float* x   = (const float*)d_in[0];
    const float* w1  = (const float*)d_in[1];
    const float* w2  = (const float*)d_in[2];
    const float* w3  = (const float*)d_in[3];
    const float* wfc = (const float*)d_in[4];
    const float* bfc = (const float*)d_in[5];
    float* out = (float*)d_out;

    char* ws = (char*)d_ws;
    unsigned* absmax = (unsigned*)(ws + WS_ABSMAX);
    int8_t* a1   = (int8_t*)(ws + WS_A1);
    int8_t* a2   = (int8_t*)(ws + WS_A2);
    int8_t* a3   = (int8_t*)(ws + WS_A3);
    int8_t* pool = (int8_t*)(ws + WS_POOL);
    int*    bf2  = (int*)(ws + WS_BF2);
    int*    bf3  = (int*)(ws + WS_BF3);
    int8_t* wfcs = (int8_t*)(ws + WS_WFC);

    k_init<<<1, 1, 0, stream>>>(absmax);

    const int npix1 = NIMG * 676;
    k_conv1_absmax<<<(npix1 + 255) / 256, 256, 0, stream>>>(x, w1, absmax);
    k_conv1_sign  <<<(npix1 + 255) / 256, 256, 0, stream>>>(x, w1, absmax, a1);

    k_pack_bfrag<16, 16><<<3, 256, 0, stream>>>(w2, bf2);
    k_pack_bfrag<16, 32><<<6, 256, 0, stream>>>(w3, bf3);
    k_pack_wfc<<<(10 * 3872 + 255) / 256, 256, 0, stream>>>(wfc, wfcs);

    k_bin_conv_wmma<16, 16, 26, 24><<<NIMG, 256, 0, stream>>>(a1, bf2, a2);
    k_bin_conv_wmma<16, 32, 24, 22><<<NIMG, 256, 0, stream>>>(a2, bf3, a3);

    const int npool = NIMG * 32 * 121;
    k_pool<<<(npool + 255) / 256, 256, 0, stream>>>(a3, pool);
    k_fc<<<(NIMG * 10 + 255) / 256, 256, 0, stream>>>(pool, wfcs, bfc, out);
}
